// MambaFeatureEncoder_68917045231760
// MI455X (gfx1250) — matrix operand order repaired and checked by reading, not compiled, not run on hardware
//
#include <hip/hip_runtime.h>

// ---------------- problem dims (from reference) ----------------
#define B_   8
#define T_   2048
#define NIN_ 128
#define DM_  256
#define DS_  16
#define DC_  4
#define DI_  512
#define L_   2
#define M_   (B_*T_)   // 16384 rows

// ---------------- CDNA5 WMMA types ----------------
typedef __attribute__((ext_vector_type(16))) __bf16          v16bf;
typedef __attribute__((ext_vector_type(8)))  float           v8f;
typedef __attribute__((ext_vector_type(8)))  unsigned short  ushort8;

union FragBF { v16bf v; ushort8 u[2]; };

__device__ __forceinline__ unsigned short f2bf(float f) {
  union { float f; unsigned u; } x; x.f = f;
  unsigned r = x.u + 0x7FFFu + ((x.u >> 16) & 1u);   // round-to-nearest-even
  return (unsigned short)(r >> 16);
}

// gfx1250 async global->LDS copy (ASYNCcnt-tracked, bypasses VGPRs).
// %0 = VGPR with LDS byte offset, %1 = 64-bit global address.
__device__ __forceinline__ void async_copy_b128(unsigned lds_off, const void* gptr) {
  asm volatile("global_load_async_to_lds_b128 %0, %1, off"
               :: "v"(lds_off), "v"((unsigned long long)(uintptr_t)gptr)
               : "memory");
}
// ASYNCcnt decrements in issue order -> waiting <=4 drains exactly the
// current tile's 4 copies while the next tile's 4 are still in flight.
__device__ __forceinline__ void wait_async_le4() {
  asm volatile("s_wait_asynccnt 0x4" ::: "memory");
}
__device__ __forceinline__ void wait_async0() {
  asm volatile("s_wait_asynccnt 0x0" ::: "memory");
}

// =====================================================================
// bf16 WMMA GEMM:  C[MxN] = A[MxK] @ Wt[NxK]^T (+bias)(+resid)
// Block tile 128x128, 256 threads (8 waves, 2x4), each wave 4x2 16x16
// subtiles -> 8 v_wmma_f32_16x16x32_bf16 per K-step of 32.
// LDS double-buffered; global->LDS via async copies overlapped with WMMA.
// =====================================================================
#define TM 128
#define TN 128
#define TK 32
#define AK 40   // padded LDS row (bf16 elems) -> 80B stride, conflict-free
#define WK 40

template<bool HB, bool HR>
__global__ __launch_bounds__(256) void k_gemm_bf16(
    const unsigned short* __restrict__ A,    // M x K  (bf16 bits)
    const unsigned short* __restrict__ Wt,   // N x K  (bf16 bits, pre-transposed)
    const float* __restrict__ bias,          // N      (used iff HB)
    const float* __restrict__ resid,         // M x N  (used iff HR)
    float* __restrict__ C, int M, int N, int K)
{
  __shared__ __align__(16) unsigned short As[2][TM * AK];
  __shared__ __align__(16) unsigned short Ws[2][TN * WK];

  const int tid   = threadIdx.x;
  const int lane  = tid & 31;
  const int wave  = tid >> 5;          // 0..7
  const int half  = lane >> 4;         // 0/1 (half-wave)
  const int l16   = lane & 15;
  const int waveM = wave >> 2;         // 0..1  (rows, 64 each)
  const int waveN = wave & 3;          // 0..3  (cols, 32 each)
  const int blockRow = blockIdx.x * TM;
  const int blockCol = blockIdx.y * TN;

  v8f acc[4][2];
#pragma unroll
  for (int i = 0; i < 4; ++i)
#pragma unroll
    for (int j = 0; j < 2; ++j)
#pragma unroll
      for (int r = 0; r < 8; ++r) acc[i][j][r] = 0.0f;

  // staging: both tiles are 128 rows x 32 cols of bf16; each thread moves
  // 16 bf16 (two b128 async copies) from A and from Wt.
  const int srow  = tid >> 1;    // 0..127
  const int shalf = tid & 1;     // which 16-elem chunk of the 32-wide row

  unsigned ldsA[2], ldsW[2];
#pragma unroll
  for (int bfi = 0; bfi < 2; ++bfi) {
    ldsA[bfi] = (unsigned)(uintptr_t)(&As[bfi][srow * AK + shalf * 16]);
    ldsW[bfi] = (unsigned)(uintptr_t)(&Ws[bfi][srow * WK + shalf * 16]);
  }
  const unsigned short* gA = A  + (size_t)(blockRow + srow) * K + shalf * 16;
  const unsigned short* gW = Wt + (size_t)(blockCol + srow) * K + shalf * 16;

  // prologue: stage tile k0=0 into buffer 0
  async_copy_b128(ldsA[0],      gA);
  async_copy_b128(ldsA[0] + 16, gA + 8);
  async_copy_b128(ldsW[0],      gW);
  async_copy_b128(ldsW[0] + 16, gW + 8);

  int cur = 0;
  for (int k0 = 0; k0 < K; k0 += TK, cur ^= 1) {
    const bool has_next = (k0 + TK) < K;
    if (has_next) {                       // DMA next tile under this tile's WMMAs
      const int nb = cur ^ 1;
      async_copy_b128(ldsA[nb],      gA + k0 + TK);
      async_copy_b128(ldsA[nb] + 16, gA + k0 + TK + 8);
      async_copy_b128(ldsW[nb],      gW + k0 + TK);
      async_copy_b128(ldsW[nb] + 16, gW + k0 + TK + 8);
      wait_async_le4();                   // drain only current tile's copies
    } else {
      wait_async0();
    }
    __syncthreads();

    // ---- build fragments per ISA 7.12.2 layouts ----
    FragBF fa[4], fb[2];
#pragma unroll
    for (int mi = 0; mi < 4; ++mi) {
      // A 16x32: lane(m=l16), elems 0..7 = K half*8.., elems 8..15 = K 16+half*8..
      const ushort8* p = (const ushort8*)(&As[cur][(waveM * 64 + mi * 16 + l16) * AK]);
      fa[mi].u[0] = p[half];
      fa[mi].u[1] = p[2 + half];
    }
#pragma unroll
    for (int ni = 0; ni < 2; ++ni) {
      // B 32x16: lane(n=l16), elems 0..7 = K half*8.., elems 8..15 = K 16+half*8..
      const ushort8* p = (const ushort8*)(&Ws[cur][(waveN * 32 + ni * 16 + l16) * WK]);
      fb[ni].u[0] = p[half];
      fb[ni].u[1] = p[2 + half];
    }
#pragma unroll
    for (int mi = 0; mi < 4; ++mi)
#pragma unroll
      for (int ni = 0; ni < 2; ++ni)
        acc[mi][ni] = __builtin_amdgcn_wmma_f32_16x16x32_bf16(
            false, fa[mi].v, false, fb[ni].v, (short)0, acc[mi][ni], false, false);
    __syncthreads();   // all waves done reading buf[cur] before it is re-staged
  }

  // ---- store: C/D 16x16 f32 layout: elem r -> m = r + 8*half, n = l16 ----
#pragma unroll
  for (int mi = 0; mi < 4; ++mi)
#pragma unroll
    for (int ni = 0; ni < 2; ++ni) {
      const int n = blockCol + waveN * 32 + ni * 16 + l16;
      float bv = 0.0f;
      if constexpr (HB) bv = bias[n];
#pragma unroll
      for (int r = 0; r < 8; ++r) {
        const int m = blockRow + waveM * 64 + mi * 16 + half * 8 + r;
        const size_t idx = (size_t)m * N + n;
        float v = acc[mi][ni][r] + bv;
        if constexpr (HR) v += resid[idx];
        C[idx] = v;
      }
    }
}

// =====================================================================
// fp32 -> bf16 convert (activation) and transpose-convert (weights)
// =====================================================================
__global__ __launch_bounds__(256) void k_cvt(const float* __restrict__ in,
                                             unsigned short* __restrict__ out, int n) {
  int i = blockIdx.x * 256 + threadIdx.x;
  if (i < n) out[i] = f2bf(in[i]);
}

__global__ __launch_bounds__(256) void k_wcvt_t(const float* __restrict__ W,   // K x N
                                                unsigned short* __restrict__ Wt, // N x K
                                                int K, int N) {
  int i = blockIdx.x * 256 + threadIdx.x;
  if (i >= K * N) return;
  int k = i / N, n = i - k * N;
  Wt[(size_t)n * K + k] = f2bf(W[i]);
}

// =====================================================================
// LayerNorm over DM=256, one block per row, bf16 output for WMMA A-side
// =====================================================================
__global__ __launch_bounds__(256) void k_layernorm(
    const float* __restrict__ h, const float* __restrict__ g,
    const float* __restrict__ bb, unsigned short* __restrict__ out)
{
  __shared__ float red[DM_];
  const int row = blockIdx.x, c = threadIdx.x;
  const float v = h[(size_t)row * DM_ + c];
  red[c] = v; __syncthreads();
  for (int o = DM_ / 2; o > 0; o >>= 1) { if (c < o) red[c] += red[c + o]; __syncthreads(); }
  const float mean = red[0] * (1.0f / DM_);
  __syncthreads();
  const float dv = v - mean;
  red[c] = dv * dv; __syncthreads();
  for (int o = DM_ / 2; o > 0; o >>= 1) { if (c < o) red[c] += red[c + o]; __syncthreads(); }
  const float rstd = rsqrtf(red[0] * (1.0f / DM_) + 1e-5f);
  out[(size_t)row * DM_ + c] = f2bf(dv * rstd * g[c] + bb[c]);
}

// =====================================================================
// depthwise causal conv (k=4) + bias + SiLU ; xp = xz[..., :DI]
// =====================================================================
__global__ __launch_bounds__(256) void k_conv_silu(
    const float* __restrict__ xz, const float* __restrict__ cw,
    const float* __restrict__ cb, float* __restrict__ xc)
{
  int g = blockIdx.x * 256 + threadIdx.x;
  if (g >= M_ * DI_) return;
  const int d   = g & (DI_ - 1);
  const int row = g >> 9;          // b*T + t
  const int t   = row & (T_ - 1);  // T_ is a power of two
  float acc = cb[d];
#pragma unroll
  for (int j = 0; j < DC_; ++j) {
    const int tt = t - (DC_ - 1) + j;
    if (tt >= 0) acc += xz[(size_t)(row - t + tt) * (2 * DI_) + d] * cw[d * DC_ + j];
  }
  const float s = 1.0f / (1.0f + __expf(-acc));
  xc[(size_t)g] = acc * s;
}

// =====================================================================
// xproj: dbc = xc @ xw (512 x 33); split into dt-raw / B / C streams
// (0.55 GFLOP total -> VALU is fine; N=33 doesn't tile for WMMA)
// =====================================================================
__global__ __launch_bounds__(64) void k_xproj(
    const float* __restrict__ xc, const float* __restrict__ xw,
    float* __restrict__ draw, float* __restrict__ Bm, float* __restrict__ Cm)
{
  const int row = blockIdx.x;
  const int col = threadIdx.x;
  if (col >= 2 * DS_ + 1) return;
  const float* xr = xc + (size_t)row * DI_;
  float acc = 0.0f;
  for (int k = 0; k < DI_; ++k) acc += xr[k] * xw[k * (2 * DS_ + 1) + col];
  if (col == 0)            draw[row] = acc;
  else if (col <= DS_)     Bm[(size_t)row * DS_ + (col - 1)]        = acc;
  else                     Cm[(size_t)row * DS_ + (col - 1 - DS_)]  = acc;
}

// =====================================================================
// selective scan: 256 blocks x 256 threads = 65536 lanes = B*DI*DS.
// thread = (s fast, d slow) so the DS=16 reduction is a half-wave
// shuffle tree (wave32). Fuses dt=softplus, dA=exp, +xc*D, *silu(z),
// writes y in bf16 for the output WMMA GEMM.
// =====================================================================
__global__ __launch_bounds__(256) void k_scan(
    const float* __restrict__ draw, const float* __restrict__ Bm,
    const float* __restrict__ Cm,   const float* __restrict__ xc,
    const float* __restrict__ xz,   const float* __restrict__ dtw,
    const float* __restrict__ dtb,  const float* __restrict__ Alog,
    const float* __restrict__ Dp,   unsigned short* __restrict__ ybf)
{
  const int s    = threadIdx.x & (DS_ - 1);
  const int dsub = threadIdx.x >> 4;
  const int db   = blockIdx.x & (DI_ / 16 - 1);
  const int b    = blockIdx.x >> 5;
  const int d    = db * 16 + dsub;

  const float Ads = -__expf(Alog[d * DS_ + s]);
  const float wd = dtw[d], bd = dtb[d], Dd = Dp[d];
  float hs = 0.0f;

  for (int t = 0; t < T_; ++t) {
    const size_t row = (size_t)b * T_ + t;
    if ((t & 15) == 0 && t + 32 < T_)
      __builtin_prefetch((const void*)(xc + (row + 32) * DI_ + d), 0, 1);

    const float dr  = draw[row];
    const float pre = dr * wd + bd;
    const float dt  = (pre > 20.0f) ? pre : log1pf(__expf(pre));
    const float bm  = Bm[row * DS_ + s];
    const float cm  = Cm[row * DS_ + s];
    const float xv  = xc[row * DI_ + d];

    hs = __expf(dt * Ads) * hs + (dt * bm) * xv;

    float part = hs * cm;
    part += __shfl_xor(part, 1, 16);
    part += __shfl_xor(part, 2, 16);
    part += __shfl_xor(part, 4, 16);
    part += __shfl_xor(part, 8, 16);

    if (s == 0) {
      const float zv  = xz[row * (2 * DI_) + DI_ + d];
      const float sig = 1.0f / (1.0f + __expf(-zv));
      const float y   = (part + xv * Dd) * (zv * sig);
      ybf[row * DI_ + d] = f2bf(y);
    }
  }
}

// =====================================================================
// final: LN of last timestep per batch + oproj (8 x 256 @ 256 x 128)
// =====================================================================
__global__ __launch_bounds__(256) void k_final(
    const float* __restrict__ h, const float* __restrict__ g,
    const float* __restrict__ bb, const float* __restrict__ ow,
    const float* __restrict__ ob, float* __restrict__ out)
{
  __shared__ float red[DM_];
  __shared__ float xn[DM_];
  const int b = blockIdx.x, c = threadIdx.x;
  const float v = h[((size_t)b * T_ + (T_ - 1)) * DM_ + c];
  red[c] = v; __syncthreads();
  for (int o = DM_ / 2; o > 0; o >>= 1) { if (c < o) red[c] += red[c + o]; __syncthreads(); }
  const float mean = red[0] * (1.0f / DM_);
  __syncthreads();
  const float dv = v - mean;
  red[c] = dv * dv; __syncthreads();
  for (int o = DM_ / 2; o > 0; o >>= 1) { if (c < o) red[c] += red[c + o]; __syncthreads(); }
  const float rstd = rsqrtf(red[0] * (1.0f / DM_) + 1e-5f);
  xn[c] = dv * rstd * g[c] + bb[c];
  __syncthreads();
  if (c < NIN_) {
    float acc = ob[c];
    for (int k = 0; k < DM_; ++k) acc += xn[k] * ow[k * NIN_ + c];
    out[b * NIN_ + c] = acc;
  }
}

// =====================================================================
// host launcher
// =====================================================================
extern "C" void kernel_launch(void* const* d_in, const int* in_sizes, int n_in,
                              void* d_out, int out_size, void* d_ws, size_t ws_size,
                              hipStream_t stream)
{
  (void)in_sizes; (void)n_in; (void)out_size; (void)ws_size;

  const float* x        = (const float*)d_in[0];
  const float* inproj_w = (const float*)d_in[1];
  const float* inproj_b = (const float*)d_in[2];
  const float* ln_g     = (const float*)d_in[3];
  const float* ln_b     = (const float*)d_in[4];
  const float* in_w     = (const float*)d_in[5];
  const float* conv_w   = (const float*)d_in[6];
  const float* conv_b   = (const float*)d_in[7];
  const float* xproj_w  = (const float*)d_in[8];
  const float* dt_w     = (const float*)d_in[9];
  const float* dt_b     = (const float*)d_in[10];
  const float* A_log    = (const float*)d_in[11];
  const float* Dp       = (const float*)d_in[12];
  const float* out_w    = (const float*)d_in[13];
  const float* onorm_g  = (const float*)d_in[14];
  const float* onorm_b  = (const float*)d_in[15];
  const float* oproj_w  = (const float*)d_in[16];
  const float* oproj_b  = (const float*)d_in[17];

  char* wsp = (char*)d_ws;
  size_t off = 0;
  auto alloc = [&](size_t bytes) -> void* {
    void* p = wsp + off;
    off = (off + bytes + 255) & ~(size_t)255;
    return p;
  };

  unsigned short* x_bf  = (unsigned short*)alloc((size_t)M_ * NIN_ * 2);
  float*          h     = (float*)         alloc((size_t)M_ * DM_ * 4);
  unsigned short* xn_bf = (unsigned short*)alloc((size_t)M_ * DM_ * 2);
  float*          xz    = (float*)         alloc((size_t)M_ * 2 * DI_ * 4);
  float*          xc    = (float*)         alloc((size_t)M_ * DI_ * 4);
  unsigned short* y_bf  = (unsigned short*)alloc((size_t)M_ * DI_ * 2);
  float*          draw  = (float*)         alloc((size_t)M_ * 4);
  float*          Bmat  = (float*)         alloc((size_t)M_ * DS_ * 4);
  float*          Cmat  = (float*)         alloc((size_t)M_ * DS_ * 4);
  unsigned short* WtIn  = (unsigned short*)alloc((size_t)DM_ * NIN_ * 2);
  unsigned short* WtL[L_];
  unsigned short* WtO[L_];
  for (int l = 0; l < L_; ++l) WtL[l] = (unsigned short*)alloc((size_t)2 * DI_ * DM_ * 2);
  for (int l = 0; l < L_; ++l) WtO[l] = (unsigned short*)alloc((size_t)DM_ * DI_ * 2);

  // ---- weight conversion (bf16, pre-transposed to N x K) ----
  k_wcvt_t<<<(NIN_ * DM_ + 255) / 256, 256, 0, stream>>>(inproj_w, WtIn, NIN_, DM_);
  for (int l = 0; l < L_; ++l)
    k_wcvt_t<<<(DM_ * 2 * DI_ + 255) / 256, 256, 0, stream>>>(in_w + (size_t)l * DM_ * 2 * DI_, WtL[l], DM_, 2 * DI_);
  for (int l = 0; l < L_; ++l)
    k_wcvt_t<<<(DI_ * DM_ + 255) / 256, 256, 0, stream>>>(out_w + (size_t)l * DI_ * DM_, WtO[l], DI_, DM_);
  k_cvt<<<(M_ * NIN_ + 255) / 256, 256, 0, stream>>>(x, x_bf, M_ * NIN_);

  // ---- h = x @ inproj_w + b ----
  k_gemm_bf16<true, false><<<dim3(M_ / TM, DM_ / TN), 256, 0, stream>>>(
      x_bf, WtIn, inproj_b, nullptr, h, M_, DM_, NIN_);

  for (int l = 0; l < L_; ++l) {
    k_layernorm<<<M_, 256, 0, stream>>>(h, ln_g + l * DM_, ln_b + l * DM_, xn_bf);
    k_gemm_bf16<false, false><<<dim3(M_ / TM, (2 * DI_) / TN), 256, 0, stream>>>(
        xn_bf, WtL[l], nullptr, nullptr, xz, M_, 2 * DI_, DM_);
    k_conv_silu<<<(M_ * DI_ + 255) / 256, 256, 0, stream>>>(
        xz, conv_w + (size_t)l * DI_ * DC_, conv_b + (size_t)l * DI_, xc);
    k_xproj<<<M_, 64, 0, stream>>>(
        xc, xproj_w + (size_t)l * DI_ * (2 * DS_ + 1), draw, Bmat, Cmat);
    k_scan<<<B_ * (DI_ / 16), 256, 0, stream>>>(
        draw, Bmat, Cmat, xc, xz, dt_w + (size_t)l * DI_, dt_b + (size_t)l * DI_,
        A_log + (size_t)l * DI_ * DS_, Dp + (size_t)l * DI_, y_bf);
    // h = y @ out_w + h   (residual fused, in place: each element RMW by one thread)
    k_gemm_bf16<false, true><<<dim3(M_ / TM, DM_ / TN), 256, 0, stream>>>(
        y_bf, WtO[l], nullptr, h, h, M_, DM_, DI_);
  }

  k_final<<<B_, 256, 0, stream>>>(h, onorm_g, onorm_b, oproj_w, oproj_b, (float*)d_out);
}
